// VICRegL_87101936763682
// MI455X (gfx1250) — matrix operand order repaired and checked by reading, not compiled, not run on hardware
//
#include <hip/hip_runtime.h>
#include <hip/hip_bf16.h>

typedef __bf16 bf16_t;
typedef __attribute__((ext_vector_type(16))) __bf16 v16bf;
typedef __attribute__((ext_vector_type(8)))  __bf16 bf16x8;
typedef __attribute__((ext_vector_type(4)))  __bf16 bf16x4;
typedef __attribute__((ext_vector_type(8)))  float  v8f;
typedef __attribute__((ext_vector_type(4)))  unsigned int u32x4;
typedef __attribute__((ext_vector_type(8)))  int  i32x8;
typedef __attribute__((ext_vector_type(4)))  int  i32x4;

#define NROW 784
#define NCH  512
#define NB   64
#define MT   49   // 784/16 column tiles

// ---------------------------------------------------------------------------
// 0) One-shot f32 -> bf16 mirror of both maps: bf[2][B][784][512]
// ---------------------------------------------------------------------------
__global__ __launch_bounds__(256) void tobf16_kernel(const float* __restrict__ m1,
                                                     const float* __restrict__ m2,
                                                     bf16_t* __restrict__ out) {
  const size_t idx  = ((size_t)blockIdx.x * 256 + threadIdx.x) * 4;
  const size_t half = (size_t)NB * NROW * NCH;
  const float* src = (idx < half) ? (m1 + idx) : (m2 + (idx - half));
  const float4 v = *(const float4*)src;
  bf16x4 o;
  o[0] = (bf16_t)v.x; o[1] = (bf16_t)v.y; o[2] = (bf16_t)v.z; o[3] = (bf16_t)v.w;
  *(bf16x4*)(out + idx) = o;
}

// ---------------------------------------------------------------------------
// 1) Row sum-of-squares (full f32 precision): sq[map][b][row]
// ---------------------------------------------------------------------------
__global__ __launch_bounds__(256) void rowsq_kernel(const float* __restrict__ m1,
                                                    const float* __restrict__ m2,
                                                    float* __restrict__ sq) {
  const int lane = threadIdx.x & 31;
  const int row  = blockIdx.x * 8 + (threadIdx.x >> 5);
  const int b    = blockIdx.y;
  const int mp   = blockIdx.z;
  if (row >= NROW) return;
  const float* src = (mp ? m2 : m1) + ((size_t)b * NROW + row) * NCH;
  float s = 0.f;
  for (int c = lane; c < NCH; c += 32) { float v = src[c]; s += v * v; }
  for (int o = 16; o; o >>= 1) s += __shfl_down(s, o);
  if (lane == 0) sq[((size_t)mp * NB + b) * NROW + row] = s;
}

// ---------------------------------------------------------------------------
// 2a) Fused feature cdist + row-min/argmin, bf16-mirror fast path.
//     A tile staged to LDS by the Tensor Data Mover (TDM) when available.
//     Block: 8 waves over 32 A-rows (2 subtiles); wave w: subtile (w&1),
//     m-tiles mt = (w>>1) + 4k.
// ---------------------------------------------------------------------------
__global__ __launch_bounds__(256) void featnn_bf16_kernel(const bf16_t* __restrict__ bfmaps,
                                                          const float* __restrict__ sq,
                                                          float* __restrict__ minval,
                                                          int*   __restrict__ nnidx) {
  const int bt  = blockIdx.x;          // 0..24 (32-row tiles)
  const int b   = blockIdx.y;
  const int dir = blockIdx.z;          // 0: A=maps1,B=maps2 ; 1: swapped
  const bf16_t* Abf = bfmaps + ((size_t)dir       * NB + b) * NROW * NCH;
  const bf16_t* Bbf = bfmaps + ((size_t)(1 - dir) * NB + b) * NROW * NCH;
  const float* sqA = sq + ((size_t)dir       * NB + b) * NROW;
  const float* sqB = sq + ((size_t)(1 - dir) * NB + b) * NROW;

  __shared__ alignas(16) bf16_t Atile[32][520];  // pitch 1040 B (TDM pad: 1 row + 4 DW)
  __shared__ float Stile[8][16][17];
  __shared__ float wval[8][16];
  __shared__ int   widx[8][16];

  const int tid = threadIdx.x;

#if defined(__gfx1250__) && __has_builtin(__builtin_amdgcn_tensor_load_to_lds)
  // --- TDM: DMA the 32x512 bf16 A tile into LDS (one wave issues, all wait) ---
  if (tid < 32) {
    const unsigned long long ga =
        (unsigned long long)(size_t)(Abf + (size_t)(bt * 32) * NCH);
    const unsigned lds_off = (unsigned)(size_t)(&Atile[0][0]);
    int rows = NROW - bt * 32; if (rows > 32) rows = 32;   // OOB rows zero-fill
    u32x4 g0;
    g0[0] = 1u;                                            // count=1 (user D#)
    g0[1] = lds_off;                                       // lds_addr
    g0[2] = (unsigned)(ga & 0xffffffffu);                  // global_addr[31:0]
    g0[3] = (unsigned)((ga >> 32) & 0x01ffffffu) | (2u << 30); // addr[56:32] | type=2
    i32x8 g1;
    g1[0] = (1 << 16) | (1 << 20) | (7 << 22) | (3 << 25); // data_size=2B, pad_en,
                                                           // pad_interval=256DW, pad=4DW
    g1[1] = (NCH & 0xffff) << 16;                          // tensor_dim0 = 512
    g1[2] = (rows & 0xffff) << 16;                         // tensor_dim1 = rows
    g1[3] = (NCH & 0xffff) << 16;                          // tile_dim0 = 512
    g1[4] = 32;                                            // tile_dim1 = 32, tile_dim2 = 0
    g1[5] = NCH;                                           // tensor_dim0_stride = 512
    g1[6] = 0;
    g1[7] = 0;
    i32x4 g2 = {0, 0, 0, 0};
    i32x4 g3 = {0, 0, 0, 0};
#if __clang_major__ >= 23
    i32x8 g4 = {0, 0, 0, 0, 0, 0, 0, 0};
    __builtin_amdgcn_tensor_load_to_lds(g0, g1, g2, g3, g4, 0);
#else
    __builtin_amdgcn_tensor_load_to_lds(g0, g1, g2, g3, 0);
#endif
    __builtin_amdgcn_s_wait_tensorcnt(0);
  }
#else
  for (int i = tid; i < 32 * NCH; i += 256) {
    const int r = i >> 9, c = i & (NCH - 1);
    const int gr = bt * 32 + r;
    Atile[r][c] = (gr < NROW) ? Abf[(size_t)gr * NCH + c] : (bf16_t)0.f;
  }
#endif
  __syncthreads();

  const int w = tid >> 5, lane = tid & 31;
  const int lr = lane & 15, hi = lane >> 4;
  const int st = w & 1;
  const int rowbase = bt * 32 + st * 16;    // multiple of 16; 784 = 49*16
  const bool rowvalid = (rowbase < NROW);

  float bestv = 3.4e38f;
  int   besti = 0;

  for (int mt = (w >> 1); mt < MT; mt += 4) {
    if (!rowvalid) continue;                // wave-uniform
    v8f acc = {};
    const int m0 = mt * 16;
    const bf16_t* Brow = Bbf + (size_t)(m0 + lr) * NCH;
    for (int kb = 0; kb < NCH; kb += 32) {
      const int ka = kb + (hi ? 16 : 0);
      const bf16x8 alo = *(const bf16x8*)&Atile[st * 16 + lr][ka];
      const bf16x8 ahi = *(const bf16x8*)&Atile[st * 16 + lr][ka + 8];
      const v16bf av = __builtin_shufflevector(alo, ahi, 0, 1, 2, 3, 4, 5, 6, 7,
                                               8, 9, 10, 11, 12, 13, 14, 15);
      const int kbB = kb + (hi ? 16 : 0);
      __builtin_prefetch(Brow + kbB + 256, 0, 0);   // global_prefetch_b8
      const v16bf bv = *(const v16bf*)(Brow + kbB);
      acc = __builtin_amdgcn_wmma_f32_16x16x32_bf16(false, av, false, bv,
                                                    (short)0, acc, false, false);
    }
#pragma unroll
    for (int i = 0; i < 8; ++i) Stile[w][i + hi * 8][lr] = acc[i];
    if (lane < 16) {
      for (int c = 0; c < 16; ++c) {
        const int m = m0 + c;
        const float t = sqB[m] - 2.f * Stile[w][lane][c];
        if (t < bestv) { bestv = t; besti = m; }
      }
    }
  }
  if (lane < 16) { wval[w][lane] = bestv; widx[w][lane] = besti; }
  __syncthreads();
  if (tid < 32) {
    const int stt = tid >> 4, r = tid & 15;
    const int grow = bt * 32 + stt * 16 + r;
    if (grow < NROW) {
      float bv = 3.4e38f; int bi = 0;
      for (int ww = stt; ww < 8; ww += 2) {
        const float v = wval[ww][r]; const int ix = widx[ww][r];
        if (v < bv || (v == bv && ix < bi)) { bv = v; bi = ix; }
      }
      float d2 = sqA[grow] + bv;
      if (d2 < 0.f) d2 = 0.f;
      const size_t o = ((size_t)dir * NB + b) * NROW + grow;
      minval[o] = d2;
      nnidx[o]  = bi;
    }
  }
}

// ---------------------------------------------------------------------------
// 2b) f32 fallback (used only if workspace cannot hold the bf16 mirror)
// ---------------------------------------------------------------------------
__global__ __launch_bounds__(256) void featnn_f32_kernel(const float* __restrict__ maps1,
                                                         const float* __restrict__ maps2,
                                                         const float* __restrict__ sq,
                                                         float* __restrict__ minval,
                                                         int*   __restrict__ nnidx) {
  const int bt  = blockIdx.x;
  const int b   = blockIdx.y;
  const int dir = blockIdx.z;
  const float* A  = dir ? maps2 : maps1;
  const float* Bm = dir ? maps1 : maps2;
  const float* sqA = sq + ((size_t)dir       * NB + b) * NROW;
  const float* sqB = sq + ((size_t)(1 - dir) * NB + b) * NROW;

  __shared__ bf16_t Atile[32][520];
  __shared__ float  Stile[8][16][17];
  __shared__ float  wval[8][16];
  __shared__ int    widx[8][16];

  const int tid = threadIdx.x;
  for (int i = tid; i < 32 * NCH; i += 256) {
    const int r = i >> 9, c = i & (NCH - 1);
    const int gr = bt * 32 + r;
    const float v = (gr < NROW) ? A[((size_t)b * NROW + gr) * NCH + c] : 0.f;
    Atile[r][c] = (bf16_t)v;
  }
  __syncthreads();

  const int w = tid >> 5, lane = tid & 31;
  const int lr = lane & 15, hi = lane >> 4;
  const int st = w & 1;
  const int rowbase = bt * 32 + st * 16;
  const bool rowvalid = (rowbase < NROW);

  float bestv = 3.4e38f;
  int   besti = 0;

  for (int mt = (w >> 1); mt < MT; mt += 4) {
    if (!rowvalid) continue;
    v8f acc = {};
    const int m0 = mt * 16;
    const float* Brow = &Bm[((size_t)b * NROW + m0 + lr) * NCH];
    for (int kb = 0; kb < NCH; kb += 32) {
      v16bf av, bv;
      const int ka = kb + (hi ? 16 : 0);
#pragma unroll
      for (int t = 0; t < 8; ++t) {
        av[t]     = Atile[st * 16 + lr][ka + t];
        av[t + 8] = Atile[st * 16 + lr][ka + 8 + t];
      }
      const int kbB = kb + (hi ? 16 : 0);
#pragma unroll
      for (int t = 0; t < 16; ++t) bv[t] = (bf16_t)Brow[kbB + t];
      acc = __builtin_amdgcn_wmma_f32_16x16x32_bf16(false, av, false, bv,
                                                    (short)0, acc, false, false);
    }
#pragma unroll
    for (int i = 0; i < 8; ++i) Stile[w][i + hi * 8][lr] = acc[i];
    if (lane < 16) {
      for (int c = 0; c < 16; ++c) {
        const int m = m0 + c;
        const float t = sqB[m] - 2.f * Stile[w][lane][c];
        if (t < bestv) { bestv = t; besti = m; }
      }
    }
  }
  if (lane < 16) { wval[w][lane] = bestv; widx[w][lane] = besti; }
  __syncthreads();
  if (tid < 32) {
    const int stt = tid >> 4, r = tid & 15;
    const int grow = bt * 32 + stt * 16 + r;
    if (grow < NROW) {
      float bv = 3.4e38f; int bi = 0;
      for (int ww = stt; ww < 8; ww += 2) {
        const float v = wval[ww][r]; const int ix = widx[ww][r];
        if (v < bv || (v == bv && ix < bi)) { bv = v; bi = ix; }
      }
      float d2 = sqA[grow] + bv;
      if (d2 < 0.f) d2 = 0.f;
      const size_t o = ((size_t)dir * NB + b) * NROW + grow;
      minval[o] = d2;
      nnidx[o]  = bi;
    }
  }
}

// ---------------------------------------------------------------------------
// 3) Location-space NN (C=2): cases 2 and 3
// ---------------------------------------------------------------------------
__global__ __launch_bounds__(256) void locnn_kernel(const float* __restrict__ loc1,
                                                    const float* __restrict__ loc2,
                                                    float* __restrict__ minval,
                                                    int*   __restrict__ nnidx) {
  __shared__ float lb[NROW * 2];
  const int b = blockIdx.y, dir = blockIdx.z;
  const float* A  = dir ? loc2 : loc1;
  const float* Bm = dir ? loc1 : loc2;
  for (int i = threadIdx.x; i < NROW * 2; i += 256)
    lb[i] = Bm[(size_t)b * NROW * 2 + i];
  __syncthreads();
  const int n = blockIdx.x * 256 + threadIdx.x;
  if (n >= NROW) return;
  const float ax = A[((size_t)b * NROW + n) * 2];
  const float ay = A[((size_t)b * NROW + n) * 2 + 1];
  float bv = 3.4e38f; int bi = 0;
  for (int m = 0; m < NROW; ++m) {
    const float dx = ax - lb[2 * m], dy = ay - lb[2 * m + 1];
    const float d = dx * dx + dy * dy;
    if (d < bv) { bv = d; bi = m; }
  }
  const size_t o = ((size_t)(2 + dir) * NB + b) * NROW + n;
  minval[o] = bv;
  nnidx[o]  = bi;
}

// ---------------------------------------------------------------------------
// 4) Stable top-M selection (rank with index tie-break == argsort semantics)
// ---------------------------------------------------------------------------
__global__ __launch_bounds__(256) void select_kernel(const float* __restrict__ minval,
                                                     const int*   __restrict__ nnidx,
                                                     const int*   __restrict__ nm,
                                                     int* __restrict__ sel_idx,
                                                     int* __restrict__ sel_nn) {
  __shared__ float v[NROW];
  __shared__ unsigned char flag[NROW];
  const int b = blockIdx.x, cs = blockIdx.y;
  const int M = nm[0];
  const float* mv = minval + ((size_t)cs * NB + b) * NROW;
  const int*   ni = nnidx  + ((size_t)cs * NB + b) * NROW;
  for (int i = threadIdx.x; i < NROW; i += 256) v[i] = mv[i];
  __syncthreads();
  for (int n = threadIdx.x; n < NROW; n += 256) {
    const float x = v[n];
    int rank = 0;
    for (int k = 0; k < NROW; ++k) {
      const float y = v[k];
      rank += (y < x) || (y == x && k < n);
    }
    flag[n] = (rank < M) ? 1 : 0;
  }
  __syncthreads();
  if (threadIdx.x == 0) {
    int pos = 0;
    int* so = sel_idx + ((size_t)cs * NB + b) * 256;
    int* sn = sel_nn  + ((size_t)cs * NB + b) * 256;
    for (int n = 0; n < NROW && pos < M; ++n)
      if (flag[n]) { so[pos] = n; sn[pos] = ni[n]; ++pos; }
  }
}

// ---------------------------------------------------------------------------
// 5) Streaming stats: repr sum, hinge(1-std) sums, batch means
// ---------------------------------------------------------------------------
__global__ __launch_bounds__(256) void stats_kernel(const float* __restrict__ m1,
                                                    const float* __restrict__ m2,
                                                    const int* __restrict__ sel_idx,
                                                    const int* __restrict__ sel_nn,
                                                    const int* __restrict__ nm,
                                                    float* __restrict__ meanbuf,
                                                    float* __restrict__ stats_part) {
  const int j = blockIdx.x, p = blockIdx.y;
  const int tid = threadIdx.x;
  float* part = stats_part + ((size_t)p * 256 + j) * 3;
  if (j >= nm[0]) { if (tid == 0) { part[0] = part[1] = part[2] = 0.f; } return; }

  const float* mx = (p & 1) ? m2 : m1;
  const float* my = (p & 1) ? m1 : m2;
  const int* si = sel_idx + (size_t)p * NB * 256;
  const int* sn = sel_nn  + (size_t)p * NB * 256;

  float repr = 0.f, hx = 0.f, hy = 0.f;
  for (int c = tid; c < NCH; c += 256) {
    float sx = 0, sxx = 0, sy = 0, syy = 0, sdd = 0;
    for (int b = 0; b < NB; ++b) {
      const int ix = si[b * 256 + j], iy = sn[b * 256 + j];
      const float x = mx[((size_t)b * NROW + ix) * NCH + c];
      const float y = my[((size_t)b * NROW + iy) * NCH + c];
      sx += x; sxx += x * x; sy += y; syy += y * y;
      const float d = x - y; sdd += d * d;
    }
    const float mxm = sx / 64.f, mym = sy / 64.f;
    meanbuf[(((size_t)0 * 4 + p) * 256 + j) * NCH + c] = mxm;
    meanbuf[(((size_t)1 * 4 + p) * 256 + j) * NCH + c] = mym;
    const float vx = (sxx - 64.f * mxm * mxm) / 63.f;
    const float vy = (syy - 64.f * mym * mym) / 63.f;
    hx += fmaxf(0.f, 1.f - sqrtf(vx + 1e-4f));
    hy += fmaxf(0.f, 1.f - sqrtf(vy + 1e-4f));
    repr += sdd;
  }
  __shared__ float r0[256], r1[256], r2[256];
  r0[tid] = repr; r1[tid] = hx; r2[tid] = hy;
  __syncthreads();
  for (int o = 128; o; o >>= 1) {
    if (tid < o) { r0[tid] += r0[tid + o]; r1[tid] += r1[tid + o]; r2[tid] += r2[tid + o]; }
    __syncthreads();
  }
  if (tid == 0) { part[0] = r0[0]; part[1] = r1[0]; part[2] = r2[0]; }
}

// ---------------------------------------------------------------------------
// 6) Covariance via 64x64 Gram trick (WMMA bf16, f32 accumulate)
// ---------------------------------------------------------------------------
__global__ __launch_bounds__(256) void gram_kernel(const float* __restrict__ m1,
                                                   const float* __restrict__ m2,
                                                   const int* __restrict__ sel_idx,
                                                   const int* __restrict__ sel_nn,
                                                   const int* __restrict__ nm,
                                                   const float* __restrict__ meanbuf,
                                                   float* __restrict__ cov_part) {
  const int j = blockIdx.x, p = blockIdx.y;
  const int tid = threadIdx.x;
  if (j >= nm[0]) { if (tid == 0) cov_part[(size_t)p * 256 + j] = 0.f; return; }

  __shared__ float X[64][68];
  __shared__ float wsum[8];
  __shared__ float dsum[64];

  const int w = tid >> 5, lane = tid & 31, lr = lane & 15, hi = lane >> 4;
  const int t0 = 2 * w, t1 = 2 * w + 1;
  const int ti0 = t0 >> 2, tj0 = t0 & 3, ti1 = t1 >> 2, tj1 = t1 & 3;

  float covacc = 0.f;
  for (int s = 0; s < 2; ++s) {
    const float* mp = ((p & 1) ^ s) ? m2 : m1;
    const int* idxarr = s ? (sel_nn + (size_t)p * NB * 256)
                          : (sel_idx + (size_t)p * NB * 256);
    const float* mean = meanbuf + (((size_t)s * 4 + p) * 256 + j) * NCH;
    v8f acc0 = {}, acc1 = {};
    float dloc = 0.f;
    for (int slab = 0; slab < 8; ++slab) {
      __syncthreads();
      const int kb = slab * 64;
      for (int i = tid; i < 64 * 64; i += 256) {
        const int r = i >> 6, c = i & 63;
        const int ridx = idxarr[r * 256 + j];
        X[r][c] = mp[((size_t)r * NROW + ridx) * NCH + kb + c] - mean[kb + c];
      }
      __syncthreads();
      if (tid < 64) {
        float dsq = 0.f;
        for (int r = 0; r < 64; ++r) { const float v = X[r][tid]; dsq += v * v; }
        dloc += dsq * dsq;
      }
      for (int ks = 0; ks < 64; ks += 32) {
        v16bf a0, b0, a1, b1;
        const int ka = ks + (hi ? 16 : 0);
#pragma unroll
        for (int t = 0; t < 8; ++t) {
          a0[t]     = (bf16_t)X[ti0 * 16 + lr][ka + t];
          a0[t + 8] = (bf16_t)X[ti0 * 16 + lr][ka + 8 + t];
          a1[t]     = (bf16_t)X[ti1 * 16 + lr][ka + t];
          a1[t + 8] = (bf16_t)X[ti1 * 16 + lr][ka + 8 + t];
        }
        const int kbb = ks + (hi ? 16 : 0);
#pragma unroll
        for (int t = 0; t < 16; ++t) {
          b0[t] = (bf16_t)X[tj0 * 16 + lr][kbb + t];
          b1[t] = (bf16_t)X[tj1 * 16 + lr][kbb + t];
        }
        acc0 = __builtin_amdgcn_wmma_f32_16x16x32_bf16(false, a0, false, b0,
                                                       (short)0, acc0, false, false);
        acc1 = __builtin_amdgcn_wmma_f32_16x16x32_bf16(false, a1, false, b1,
                                                       (short)0, acc1, false, false);
      }
    }
    float fs = 0.f;
#pragma unroll
    for (int i = 0; i < 8; ++i) fs += acc0[i] * acc0[i] + acc1[i] * acc1[i];
    for (int o = 16; o; o >>= 1) fs += __shfl_down(fs, o);
    __syncthreads();
    if (lane == 0) wsum[w] = fs;
    if (tid < 64) dsum[tid] = dloc;
    __syncthreads();
    if (tid == 0) {
      float frob = 0.f; for (int i = 0; i < 8;  ++i) frob += wsum[i];
      float dsq  = 0.f; for (int i = 0; i < 64; ++i) dsq  += dsum[i];
      covacc += (frob - dsq);
    }
  }
  if (tid == 0) {
    const float inv = 1.f / (63.f * 63.f * (float)NCH * 2.f);
    cov_part[(size_t)p * 256 + j] = covacc * inv;
  }
}

// ---------------------------------------------------------------------------
// 7) Final deterministic reduction -> scalar loss
// ---------------------------------------------------------------------------
__global__ __launch_bounds__(256) void final_kernel(const float* __restrict__ stats_part,
                                                    const float* __restrict__ cov_part,
                                                    const int* __restrict__ nm,
                                                    float* __restrict__ out) {
  __shared__ float rr[256], rh[256], rc[256];
  const int tid = threadIdx.x;
  float sr = 0.f, sh = 0.f, sc = 0.f;
  for (int i = tid; i < 1024; i += 256) {
    sr += stats_part[i * 3 + 0];
    sh += stats_part[i * 3 + 1] + stats_part[i * 3 + 2];
    sc += cov_part[i];
  }
  rr[tid] = sr; rh[tid] = sh; rc[tid] = sc;
  __syncthreads();
  for (int o = 128; o; o >>= 1) {
    if (tid < o) { rr[tid] += rr[tid + o]; rh[tid] += rh[tid + o]; rc[tid] += rc[tid + o]; }
    __syncthreads();
  }
  if (tid == 0) {
    const float Mf = (float)nm[0];
    const float inv_l = 25.f * rr[0] / (64.f * Mf * (float)NCH);
    const float var_l = 25.f * rh[0] / (2.f * Mf * (float)NCH);
    const float cov_l = 1.f * rc[0] / Mf;
    out[0] = inv_l + var_l + cov_l;
  }
}

// ---------------------------------------------------------------------------
extern "C" void kernel_launch(void* const* d_in, const int* in_sizes, int n_in,
                              void* d_out, int out_size, void* d_ws, size_t ws_size,
                              hipStream_t stream) {
  const float* maps1 = (const float*)d_in[0];
  const float* maps2 = (const float*)d_in[1];
  const float* loc1  = (const float*)d_in[2];
  const float* loc2  = (const float*)d_in[3];
  const int*   nm    = (const int*)d_in[4];
  (void)in_sizes; (void)n_in; (void)out_size;

  char* ws = (char*)d_ws;
  size_t off = 0;
  auto alloc = [&](size_t bytes) {
    size_t o = off;
    off = (off + bytes + 255) & ~(size_t)255;
    return o;
  };
  float* sq      = (float*)(ws + alloc(2UL * NB * NROW * 4));
  float* minval  = (float*)(ws + alloc(4UL * NB * NROW * 4));
  int*   nnidx   = (int*)  (ws + alloc(4UL * NB * NROW * 4));
  int*   sel_idx = (int*)  (ws + alloc(4UL * NB * 256 * 4));
  int*   sel_nn  = (int*)  (ws + alloc(4UL * NB * 256 * 4));
  float* meanbuf = (float*)(ws + alloc(2UL * 4 * 256 * NCH * 4));
  float* spart   = (float*)(ws + alloc(1024UL * 3 * 4));
  float* cpart   = (float*)(ws + alloc(1024UL * 4));
  const size_t bfbytes = 2UL * NB * NROW * NCH * sizeof(bf16_t);
  bf16_t* bfbuf  = (bf16_t*)(ws + alloc(bfbytes));
  const bool use_bf16 = (off <= ws_size);

  rowsq_kernel<<<dim3(98, NB, 2), 256, 0, stream>>>(maps1, maps2, sq);
  if (use_bf16) {
    const size_t nquads = 2UL * NB * NROW * NCH / 4;       // 12,845,056
    tobf16_kernel<<<dim3((unsigned)(nquads / 256)), 256, 0, stream>>>(maps1, maps2, bfbuf);
    featnn_bf16_kernel<<<dim3(25, NB, 2), 256, 0, stream>>>(bfbuf, sq, minval, nnidx);
  } else {
    featnn_f32_kernel<<<dim3(25, NB, 2), 256, 0, stream>>>(maps1, maps2, sq, minval, nnidx);
  }
  locnn_kernel <<<dim3(4,  NB, 2), 256, 0, stream>>>(loc1, loc2, minval, nnidx);
  select_kernel<<<dim3(NB, 4),     256, 0, stream>>>(minval, nnidx, nm, sel_idx, sel_nn);
  stats_kernel <<<dim3(256, 4),    256, 0, stream>>>(maps1, maps2, sel_idx, sel_nn, nm, meanbuf, spart);
  gram_kernel  <<<dim3(256, 4),    256, 0, stream>>>(maps1, maps2, sel_idx, sel_nn, nm, meanbuf, cpart);
  final_kernel <<<1, 256, 0, stream>>>(spart, cpart, nm, (float*)d_out);
}
